// SwitchFeedForward_49950469652765
// MI455X (gfx1250) — compile-verified, matrix operand order checked by reading, not run on hardware
//
#include <hip/hip_runtime.h>
#include <hip/hip_bf16.h>
#include <math.h>

typedef __attribute__((ext_vector_type(16))) __bf16 v16bf;
typedef __attribute__((ext_vector_type(4)))  __bf16 v4bf;
typedef __attribute__((ext_vector_type(8)))  float  v8f;

#define T_TOK 4096
#define DDIM  1024
#define FDIM  4096
#define NEXP  8
#define MT    64            // tokens per FFN block
#define FC    512           // F-chunk held in LDS per iteration
#define XSS   (DDIM + 8)    // X tile LDS row stride (bf16): +16B/row -> conflict-free frags
#define HSS   (FC + 8)      // H chunk LDS row stride (bf16)

// gelu(tanh approx) == x * sigmoid(2*0.79788456*(x + 0.044715 x^3)); branch-free,
// sigmoid via v_exp_f32 + v_rcp_f32 (no IEEE div sequence).
__device__ __forceinline__ float gelu_fast(float x) {
  float u = x + 0.044715f * x * x * x;
  float z = 1.5957691216057308f * u;
  float s = __builtin_amdgcn_rcpf(1.0f + __expf(-z));
  return x * s;
}

// 16xK (K=32 bf16) fragment with contiguous-K storage; works for LDS A-tiles and
// transposed global B-tiles. CDNA5 16-bit layout: lanes 0-15 hold row rc, K 0-7 /
// 16-23; lanes 16-31 hold K 8-15 / 24-31. Two 16B contiguous runs -> b128 loads.
__device__ __forceinline__ v16bf load_frag(const __bf16* base, int stride, int k0, int lane) {
  const int hi = lane >> 4, rc = lane & 15;
  const __bf16* p = base + (size_t)rc * stride + k0 + hi * 8;
  v16bf f;
#pragma unroll
  for (int i = 0; i < 8; ++i) { f[i] = p[i]; f[8 + i] = p[16 + i]; }
  return f;
}

#define WMMA_BF16(acc, a, b) \
  acc = __builtin_amdgcn_wmma_f32_16x16x32_bf16(false, a, false, b, (short)0, acc, false, false)

// ---------------- transpose + f32->bf16 convert: src[R,C] -> dst[C,R] per expert ----------------
__global__ __launch_bounds__(256) void transpose_cvt_kernel(
    const float* __restrict__ src, __bf16* __restrict__ dst, int R, int C) {
  __shared__ float tileS[32][33];
  const int e = blockIdx.z;
  const float* s = src + (size_t)e * R * C;
  __bf16*      d = dst + (size_t)e * R * C;
  const int c0 = blockIdx.x * 32, r0 = blockIdx.y * 32;
  const int tx = threadIdx.x & 31, ty = threadIdx.x >> 5;   // 8 rows per pass
#pragma unroll
  for (int i = 0; i < 4; ++i)
    tileS[ty + i * 8][tx] = s[(size_t)(r0 + ty + i * 8) * C + c0 + tx];
  __syncthreads();
#pragma unroll
  for (int i = 0; i < 4; ++i)
    d[(size_t)(c0 + ty + i * 8) * R + r0 + tx] = (__bf16)tileS[tx][ty + i * 8];
}

// ---------------- gating: one wave per token ----------------
__global__ __launch_bounds__(256) void gate_kernel(
    const float* __restrict__ x, const float* __restrict__ Wr,
    float* __restrict__ glog_out, int* __restrict__ routes,
    float* __restrict__ pmax, float* __restrict__ probs) {
  const int t    = blockIdx.x * 8 + (threadIdx.x >> 5);
  const int lane = threadIdx.x & 31;
  const float* xr = x + (size_t)t * DDIM;
  float acc[NEXP];
#pragma unroll
  for (int e = 0; e < NEXP; ++e) acc[e] = 0.0f;
#pragma unroll 4
  for (int i = 0; i < 32; ++i) {
    int d = i * 32 + lane;
    float xv = xr[d];
    const float* wr = Wr + (size_t)d * NEXP;
#pragma unroll
    for (int e = 0; e < NEXP; ++e) acc[e] += xv * wr[e];
  }
#pragma unroll
  for (int off = 16; off >= 1; off >>= 1) {
#pragma unroll
    for (int e = 0; e < NEXP; ++e) acc[e] += __shfl_xor(acc[e], off, 32);
  }
  float mx = acc[0]; int am = 0;
#pragma unroll
  for (int e = 1; e < NEXP; ++e) { if (acc[e] > mx) { mx = acc[e]; am = e; } }
  float p[NEXP], den = 0.0f;
#pragma unroll
  for (int e = 0; e < NEXP; ++e) { p[e] = expf(acc[e] - mx); den += p[e]; }
  const float inv = 1.0f / den;     // p_max = exp(0)/den
  if (lane < NEXP) {
    glog_out[(size_t)t * NEXP + lane] = acc[lane];
    probs[(size_t)t * NEXP + lane]    = p[lane] * inv;
  }
  if (lane == 0) { routes[t] = am; pmax[t] = inv; }
}

// ---------------- deterministic reductions for counts & prob sums ----------------
__global__ __launch_bounds__(256) void reduce_kernel(
    const int* __restrict__ routes, const float* __restrict__ probs,
    float* __restrict__ counts_out, float* __restrict__ psum_out) {
  __shared__ float sc[256], sp[256];
  const int e = blockIdx.x;
  float c = 0.0f, s = 0.0f;
  for (int t = threadIdx.x; t < T_TOK; t += 256) {
    c += (routes[t] == e) ? 1.0f : 0.0f;
    s += probs[(size_t)t * NEXP + e];
  }
  sc[threadIdx.x] = c; sp[threadIdx.x] = s;
  __syncthreads();
  for (int o = 128; o > 0; o >>= 1) {
    if (threadIdx.x < o) {
      sc[threadIdx.x] += sc[threadIdx.x + o];
      sp[threadIdx.x] += sp[threadIdx.x + o];
    }
    __syncthreads();
  }
  if (threadIdx.x == 0) { counts_out[e] = sc[0]; psum_out[e] = sp[0]; }
}

// ---------------- build per-expert token lists ----------------
__global__ void init_kernel(int* cnt) { if (threadIdx.x < NEXP) cnt[threadIdx.x] = 0; }

__global__ void scatter_kernel(const int* __restrict__ routes, int* cnt, int* list) {
  int t = blockIdx.x * 256 + threadIdx.x;
  if (t < T_TOK) {
    int e = routes[t];
    int i = atomicAdd(&cnt[e], 1);
    list[e * T_TOK + i] = t;
  }
}

// ---- fused expert FFN: out = (gelu(X@W1+b1) @ W2 + b2) * p_max, 64-token tiles ----
// 512 threads = 16 waves. Weights pre-transposed bf16 (K contiguous). Phase-2
// accumulators live in registers across F-chunks; H never leaves LDS.
__global__ __launch_bounds__(512) void moe_ffn_kernel(
    const float* __restrict__ x,
    const __bf16* __restrict__ W1t,   // [E, F, D] bf16
    const __bf16* __restrict__ W2t,   // [E, D, F] bf16
    const float* __restrict__ b1, const float* __restrict__ b2,
    const int* __restrict__ cnt, const int* __restrict__ list,
    const float* __restrict__ pmax, float* __restrict__ out) {
  __shared__ __bf16 Xs[MT * XSS];   // 64 x 1024 bf16 (pad 8)  ~132 KB
  __shared__ __bf16 Hs[MT * HSS];   // 64 x  512 bf16 (pad 8)  ~65 KB
  __shared__ int    toks[MT];
  __shared__ float  pms[MT];

  const int e    = blockIdx.x / (T_TOK / MT);
  const int tile = blockIdx.x % (T_TOK / MT);
  const int n_e  = cnt[e];
  if (tile * MT >= n_e) return;          // uniform exit before any barrier

  const int tid  = threadIdx.x;
  const int lane = tid & 31;
  const int wave = tid >> 5;
  const int hi   = lane >> 4;
  const int c    = lane & 15;

  if (tid < MT) {
    int idx = tile * MT + tid;
    if (idx < n_e) {
      int tk = list[e * T_TOK + idx];
      toks[tid] = tk;
      pms[tid]  = pmax[tk];
    } else {
      toks[tid] = -1;
      pms[tid]  = 0.0f;
    }
  }
  __syncthreads();

  // stage X tile (f32 -> bf16), 4 elements per thread per pass
  for (int i = tid * 4; i < MT * DDIM; i += 512 * 4) {
    int m = i >> 10, d = i & (DDIM - 1);
    int tk = toks[m];
    int src = (tk >= 0) ? tk : 0;
    const float4 v = *(const float4*)(x + (size_t)src * DDIM + d);
    v4bf h;
    h[0] = (__bf16)v.x; h[1] = (__bf16)v.y; h[2] = (__bf16)v.z; h[3] = (__bf16)v.w;
    *(v4bf*)(Xs + m * XSS + d) = h;
  }
  __syncthreads();

  const __bf16* W1e = W1t + (size_t)e * FDIM * DDIM;   // [F, D]
  const __bf16* W2e = W2t + (size_t)e * DDIM * FDIM;   // [D, F]
  const float*  b1e = b1 + (size_t)e * FDIM;
  const float*  b2e = b2 + (size_t)e * DDIM;

  v8f oacc[16];                          // 4 N-tiles x 4 M-subtiles, persistent
#pragma unroll
  for (int i = 0; i < 16; ++i) oacc[i] = (v8f){};

  for (int ch = 0; ch < FDIM / FC; ++ch) {
    const int f0 = ch * FC;

    // -------- Phase 1: H_chunk = gelu(X @ W1[:, f0:f0+FC] + b1) --------
#pragma unroll
    for (int t = 0; t < 2; ++t) {        // wave owns 2 N-tiles of the chunk
      const int fl = wave * 32 + t * 16; // chunk-local column base
      const int fg = f0 + fl;            // global column base
      const __bf16* Btile = W1e + (size_t)fg * DDIM;  // rows = output cols, K contiguous
      v8f hacc[4];
#pragma unroll
      for (int m = 0; m < 4; ++m) hacc[m] = (v8f){};
      for (int k0 = 0; k0 < DDIM; k0 += 32) {
        v16bf bfrag = load_frag(Btile, DDIM, k0, lane);
#pragma unroll
        for (int m = 0; m < 4; ++m) {    // B reused by 4 M-subtiles
          v16bf afrag = load_frag(Xs + m * 16 * XSS, XSS, k0, lane);
          WMMA_BF16(hacc[m], afrag, bfrag);
        }
      }
      const float bb = b1e[fg + c];
#pragma unroll
      for (int m = 0; m < 4; ++m) {
#pragma unroll
        for (int r = 0; r < 8; ++r) {    // D layout: VGPR r -> row r+8*hi, col c
          int row = m * 16 + r + 8 * hi;
          Hs[row * HSS + fl + c] = (__bf16)gelu_fast(hacc[m][r] + bb);
        }
      }
    }
    __syncthreads();

    // -------- Phase 2: OUT += H_chunk @ W2[f0:f0+FC, :] --------
    for (int k0 = 0; k0 < FC; k0 += 32) {
      v16bf a0 = load_frag(Hs + 0 * 16 * HSS, HSS, k0, lane);
      v16bf a1 = load_frag(Hs + 1 * 16 * HSS, HSS, k0, lane);
      v16bf a2 = load_frag(Hs + 2 * 16 * HSS, HSS, k0, lane);
      v16bf a3 = load_frag(Hs + 3 * 16 * HSS, HSS, k0, lane);
#pragma unroll
      for (int n = 0; n < 4; ++n) {
        const int d0 = wave * 64 + n * 16;
        v16bf bfrag = load_frag(W2e + (size_t)d0 * FDIM, FDIM, f0 + k0, lane);
        WMMA_BF16(oacc[n * 4 + 0], a0, bfrag);
        WMMA_BF16(oacc[n * 4 + 1], a1, bfrag);
        WMMA_BF16(oacc[n * 4 + 2], a2, bfrag);
        WMMA_BF16(oacc[n * 4 + 3], a3, bfrag);
      }
    }
    __syncthreads();                     // before next phase-1 overwrites Hs
  }

  // -------- epilogue: add b2, scale by p_max, scatter to token rows --------
#pragma unroll
  for (int n = 0; n < 4; ++n) {
    const int d0 = wave * 64 + n * 16;
    const float bb = b2e[d0 + c];
#pragma unroll
    for (int m = 0; m < 4; ++m) {
#pragma unroll
      for (int r = 0; r < 8; ++r) {
        int row = m * 16 + r + 8 * hi;
        int tk = toks[row];
        if (tk >= 0) {
          out[(size_t)tk * DDIM + d0 + c] = (oacc[n * 4 + m][r] + bb) * pms[row];
        }
      }
    }
  }
}

extern "C" void kernel_launch(void* const* d_in, const int* in_sizes, int n_in,
                              void* d_out, int out_size, void* d_ws, size_t ws_size,
                              hipStream_t stream) {
  (void)in_sizes; (void)n_in; (void)out_size; (void)ws_size;
  const float* x  = (const float*)d_in[0];   // [T, D]
  const float* Wr = (const float*)d_in[1];   // [D, E]
  const float* W1 = (const float*)d_in[2];   // [E, D, F]
  const float* b1 = (const float*)d_in[3];   // [E, F]
  const float* W2 = (const float*)d_in[4];   // [E, F, D]
  const float* b2 = (const float*)d_in[5];   // [E, D]

  float* out        = (float*)d_out;
  float* final_out  = out;                                // T*D
  float* counts_out = out + (size_t)T_TOK * DDIM;         // E
  float* psum_out   = counts_out + NEXP;                  // E
  float* glog_out   = psum_out + NEXP;                    // T*E

  char* base = (char*)d_ws;
  int*   routes = (int*)(base);                           // 16 KB
  float* pmax   = (float*)(base + 16384);                 // 16 KB
  float* probs  = (float*)(base + 32768);                 // 128 KB
  int*   cnt    = (int*)(base + 163840);                  // 32 B
  int*   list   = (int*)(base + 163872);                  // 128 KB (ends 294944)
  __bf16* W1t   = (__bf16*)(base + 295168);               // [E,F,D] bf16, 64 MB
  __bf16* W2t   = W1t + (size_t)NEXP * FDIM * DDIM;       // [E,D,F] bf16, 64 MB

  // one-shot weight transpose+convert (deterministic, per call)
  hipLaunchKernelGGL(transpose_cvt_kernel, dim3(FDIM / 32, DDIM / 32, NEXP), dim3(256), 0, stream,
                     W1, W1t, DDIM, FDIM);
  hipLaunchKernelGGL(transpose_cvt_kernel, dim3(DDIM / 32, FDIM / 32, NEXP), dim3(256), 0, stream,
                     W2, W2t, FDIM, DDIM);

  hipLaunchKernelGGL(init_kernel,    dim3(1),            dim3(32),  0, stream, cnt);
  hipLaunchKernelGGL(gate_kernel,    dim3(T_TOK / 8),    dim3(256), 0, stream,
                     x, Wr, glog_out, routes, pmax, probs);
  hipLaunchKernelGGL(reduce_kernel,  dim3(NEXP),         dim3(256), 0, stream,
                     routes, probs, counts_out, psum_out);
  hipLaunchKernelGGL(scatter_kernel, dim3(T_TOK / 256),  dim3(256), 0, stream,
                     routes, cnt, list);
  hipLaunchKernelGGL(moe_ffn_kernel, dim3(NEXP * (T_TOK / MT)), dim3(512), 0, stream,
                     x, W1t, W2t, b1, b2, cnt, list, pmax, final_out);
}
